// CoTLayer_24910810317194
// MI455X (gfx1250) — compile-verified, matrix operand order checked by reading, not run on hardware
//
#include <hip/hip_runtime.h>

typedef __bf16 bf16_t;
typedef bf16_t v8bf  __attribute__((ext_vector_type(8)));
typedef bf16_t v16bf __attribute__((ext_vector_type(16)));
typedef float  v8f   __attribute__((ext_vector_type(8)));

#define B_  2
#define S_  1024
#define R_  4
#define SC_ 4096
#define D_  1024
#define H_  16
#define HD_ 64
#define FF_ 4096

// ---------- helpers ----------

static __device__ __forceinline__ v8f vzero() {
  v8f z;
#pragma unroll
  for (int i = 0; i < 8; ++i) z[i] = 0.f;
  return z;
}

// Load one 16-bit WMMA A/B fragment (K-chunk of 32) for this lane.
// base points at the lane's row start for this K-chunk (16B aligned).
// CDNA5 layout: lane group g = lane/16 holds K = {8g..8g+7, 16+8g..16+8g+7}.
static __device__ __forceinline__ v16bf load_frag(const bf16_t* base, int g) {
  v8bf lo = *(const v8bf*)(base + 8 * g);
  v8bf hi = *(const v8bf*)(base + 16 + 8 * g);
  return __builtin_shufflevector(lo, hi, 0, 1, 2, 3, 4, 5, 6, 7,
                                 8, 9, 10, 11, 12, 13, 14, 15);
}

static __device__ __forceinline__ v8f wmma_bf16(v16bf a, v16bf b, v8f c) {
  return __builtin_amdgcn_wmma_f32_16x16x32_bf16(false, a, false, b,
                                                 (short)0, c, false, false);
}

// Async DMA of one 16B chunk: global -> LDS (CDNA5 async path, ASYNCcnt).
static __device__ __forceinline__ void async_copy16(const bf16_t* gptr,
                                                    bf16_t* lptr) {
  unsigned lds_off = (unsigned)(size_t)(void*)lptr;  // LDS aperture low 32 bits
  unsigned long long ga = (unsigned long long)(size_t)gptr;
  asm volatile("global_load_async_to_lds_b128 %0, %1, off"
               :: "v"(lds_off), "v"(ga)
               : "memory");
}

static __device__ __forceinline__ void wait_async0() {
  asm volatile("s_wait_asynccnt 0x0" ::: "memory");
}

// ---------- weight transpose + convert: W f32 [K][N] -> Wt bf16 [N][K] ----------

__global__ __launch_bounds__(256) void transpose_w(const float* __restrict__ W,
                                                   bf16_t* __restrict__ Wt,
                                                   int K, int N) {
  __shared__ float tile[32][33];
  const int k0 = blockIdx.x * 32, n0 = blockIdx.y * 32;
  const int tx = threadIdx.x & 31, ty = threadIdx.x >> 5;  // 32 x 8
#pragma unroll
  for (int i = ty; i < 32; i += 8)
    tile[i][tx] = W[(size_t)(k0 + i) * N + n0 + tx];
  __syncthreads();
#pragma unroll
  for (int i = ty; i < 32; i += 8)
    Wt[(size_t)(n0 + i) * K + k0 + tx] = (bf16_t)tile[tx][i];
}

// ---------- layernorm (one 256-thread block per row of D=1024), bf16 out ----------

__global__ __launch_bounds__(256) void layernorm_k(const float* __restrict__ x,
                                                   const float* __restrict__ gam,
                                                   const float* __restrict__ bet,
                                                   bf16_t* __restrict__ out) {
  const int row = blockIdx.x;
  const float* xr = x + (size_t)row * D_;
  float s = 0.f, s2 = 0.f;
  for (int i = threadIdx.x; i < D_; i += 256) {
    float v = xr[i];
    s += v;
    s2 += v * v;
  }
#pragma unroll
  for (int m = 1; m < 32; m <<= 1) {
    s += __shfl_xor(s, m, 32);
    s2 += __shfl_xor(s2, m, 32);
  }
  __shared__ float ss[8], ss2[8];
  const int w = threadIdx.x >> 5, ln = threadIdx.x & 31;
  if (ln == 0) { ss[w] = s; ss2[w] = s2; }
  __syncthreads();
  if (threadIdx.x == 0) {
    float t = 0.f, t2 = 0.f;
#pragma unroll
    for (int i = 0; i < 8; ++i) { t += ss[i]; t2 += ss2[i]; }
    ss[0] = t; ss2[0] = t2;
  }
  __syncthreads();
  const float mean = ss[0] * (1.f / D_);
  const float var = ss2[0] * (1.f / D_) - mean * mean;
  const float rstd = rsqrtf(var + 1e-5f);
  for (int i = threadIdx.x; i < D_; i += 256) {
    float v = (xr[i] - mean) * rstd * gam[i] + bet[i];
    out[(size_t)row * D_ + i] = (bf16_t)v;
  }
}

// ---------- f32 -> bf16 convert ----------

__global__ __launch_bounds__(256) void convert_bf16(const float* __restrict__ in,
                                                    bf16_t* __restrict__ out,
                                                    size_t n) {
  size_t i = (size_t)blockIdx.x * blockDim.x + threadIdx.x;
  const size_t stride = (size_t)gridDim.x * blockDim.x;
  for (; i < n; i += stride) out[i] = (bf16_t)in[i];
}

// ---------- WMMA GEMM: C[M][N] = A_bf16[M][K] @ (Bt[N][K])^T + bias ----------
// 256 threads = 8 waves. 128x128 block tile; wave = 32(M) x 64(N).
// MODE 0: bf16 out, head-split [b][h][s][64]             (Q/K projections)
// MODE 1: bf16 out, head-split transposed [b][h][64][s]  (V projection)
// MODE 2: f32 out row-major + residual                   (attn out proj + query)
// MODE 3: bf16 out row-major, exact GELU                 (FFN up)
// MODE 4: f32 out row-major + residual                   (FFN down + x)

template <int MODE>
__global__ __launch_bounds__(256) void gemm_nt(const bf16_t* __restrict__ A,
                                               const bf16_t* __restrict__ Bt,
                                               const float* __restrict__ bias,
                                               const float* __restrict__ res,
                                               void* __restrict__ outp,
                                               int M, int N, int K, int Ssz) {
  const int wave = threadIdx.x >> 5, lane = threadIdx.x & 31;
  const int g = lane >> 4, ln = lane & 15;
  const int wm = wave & 3, wn = wave >> 2;
  const int m0 = blockIdx.x * 128 + wm * 32;
  const int n0 = blockIdx.y * 128 + wn * 64;

  v8f acc[2][4];
#pragma unroll
  for (int i = 0; i < 2; ++i)
#pragma unroll
    for (int j = 0; j < 4; ++j) acc[i][j] = vzero();

  for (int k0 = 0; k0 < K; k0 += 32) {
    v16bf af[2], bfr[4];
#pragma unroll
    for (int i = 0; i < 2; ++i)
      af[i] = load_frag(A + (size_t)(m0 + i * 16 + ln) * K + k0, g);
#pragma unroll
    for (int j = 0; j < 4; ++j)
      bfr[j] = load_frag(Bt + (size_t)(n0 + j * 16 + ln) * K + k0, g);
#pragma unroll
    for (int i = 0; i < 2; ++i)
#pragma unroll
      for (int j = 0; j < 4; ++j)
        acc[i][j] = wmma_bf16(af[i], bfr[j], acc[i][j]);
  }

#pragma unroll
  for (int i = 0; i < 2; ++i) {
#pragma unroll
    for (int j = 0; j < 4; ++j) {
#pragma unroll
      for (int r = 0; r < 8; ++r) {
        const int mg = m0 + i * 16 + 8 * g + r;
        const int ng = n0 + j * 16 + ln;
        float v = acc[i][j][r] + bias[ng];
        if constexpr (MODE == 0) {
          const int bb = mg / Ssz, sidx = mg % Ssz;
          const int hh = ng >> 6, hd = ng & 63;
          ((bf16_t*)outp)[(((size_t)bb * H_ + hh) * Ssz + sidx) * HD_ + hd] =
              (bf16_t)v;
        } else if constexpr (MODE == 1) {
          const int bb = mg / Ssz, sidx = mg % Ssz;
          const int hh = ng >> 6, hd = ng & 63;
          ((bf16_t*)outp)[(((size_t)bb * H_ + hh) * HD_ + hd) * Ssz + sidx] =
              (bf16_t)v;
        } else if constexpr (MODE == 2) {
          ((float*)outp)[(size_t)mg * N + ng] = v + res[(size_t)mg * N + ng];
        } else if constexpr (MODE == 3) {
          float ge = 0.5f * v * (1.f + erff(v * 0.70710678f));
          ((bf16_t*)outp)[(size_t)mg * N + ng] = (bf16_t)ge;
        } else {
          ((float*)outp)[(size_t)mg * N + ng] = v + res[(size_t)mg * N + ng];
        }
      }
    }
  }
}

// ---------- flash attention, causal-repeat mask, async double-buffered K/V ----------
// grid: (B*H) * (S/64) blocks, 128 threads = 4 waves.
// Each wave owns one 16-row query tile of a causal-aligned 64-row block.
// All waves consume the SAME 64-key tiles (tile granularity == block span),
// so K (64x64, row-major) and V (64hd x 64key) are staged cooperatively in
// LDS via async-to-LDS DMA, double buffered: next tile's DMA overlaps this
// tile's 16 WMMAs + online softmax.

__global__ __launch_bounds__(128) void attn_kernel(const bf16_t* __restrict__ qhd,
                                                   const bf16_t* __restrict__ kt,
                                                   const bf16_t* __restrict__ vt,
                                                   bf16_t* __restrict__ obuf) {
  __shared__ __align__(16) bf16_t smem[2][2][64 * 64];  // [buf][K/V] 32 KB
  __shared__ __align__(16) bf16_t ptile[4][16][64];     // per-wave P staging 8 KB

  const int tid = threadIdx.x;
  const int wave = tid >> 5, lane = tid & 31;
  const int g = lane >> 4, ln = lane & 15;
  const int qblocks = S_ / 64;
  const int bh = blockIdx.x / qblocks;
  const int qb = blockIdx.x % qblocks;
  const int q0 = qb * 64 + wave * 16;

  const bf16_t* qbase = qhd + ((size_t)bh * S_ + q0) * HD_;
  const bf16_t* kbase = kt + (size_t)bh * SC_ * HD_;
  const bf16_t* vbase = vt + (size_t)bh * HD_ * SC_;

  const v16bf qf0 = load_frag(qbase + (size_t)ln * HD_, g);
  const v16bf qf1 = load_frag(qbase + (size_t)ln * HD_ + 32, g);

  v8f o[4];
#pragma unroll
  for (int t = 0; t < 4; ++t) o[t] = vzero();
  float mrow[8], lrow[8];
#pragma unroll
  for (int r = 0; r < 8; ++r) { mrow[r] = -3.0e38f; lrow[r] = 0.f; }

  const float scale = 0.125f;  // 1/sqrt(64)

  // tile schedule: rb in 0..R-1, kp in 0..qb*64 step 64 (causal bound);
  // rb/kp tracked incrementally (no runtime division in the hot loop).
  const int kpmax = (qb + 1) * 64;
  const int ntiles = R_ * (qb + 1);

  // issue async stage of the tile starting at global key index kj
  auto stage = [&](int kj, int buf) {
    bf16_t* Kt = smem[buf][0];
    bf16_t* Vt = smem[buf][1];
    // 64 rows x 64 halves = 512 chunks of 16B each; 4 K-chunks + 4 V-chunks/thread
#pragma unroll
    for (int i = 0; i < 4; ++i) {
      const int c = tid + i * 128;  // 0..511
      const int row = c >> 3, col = (c & 7) * 8;
      async_copy16(kbase + (size_t)(kj + row) * HD_ + col, Kt + row * 64 + col);
      async_copy16(vbase + (size_t)row * SC_ + kj + col, Vt + row * 64 + col);
    }
  };

  stage(0, 0);
  int cur = 0;
  int rb = 0, kp = 0;

  for (int ti = 0; ti < ntiles; ++ti) {
    wait_async0();     // this wave's share of buf[cur] has landed
    __syncthreads();   // everyone's share has landed

    // next tile coordinates (incremental)
    int rb_n = rb, kp_n = kp + 64;
    if (kp_n == kpmax) { kp_n = 0; rb_n = rb + 1; }
    if (ti + 1 < ntiles) stage(rb_n * S_ + kp_n, cur ^ 1);

    const bf16_t* Kt = smem[cur][0];
    const bf16_t* Vt = smem[cur][1];

    // scores: cluster all 8 K-fragment LDS loads, then 8 WMMAs
    v16bf kf[4][2];
#pragma unroll
    for (int t = 0; t < 4; ++t) {
      const bf16_t* krow = Kt + (t * 16 + ln) * 64;
      kf[t][0] = load_frag(krow, g);
      kf[t][1] = load_frag(krow + 32, g);
    }
    v8f s[4];
#pragma unroll
    for (int t = 0; t < 4; ++t) {
      s[t] = wmma_bf16(qf0, kf[t][0], vzero());
      s[t] = wmma_bf16(qf1, kf[t][1], s[t]);
    }

    // online softmax over the 64-key tile (one max + one sum butterfly / row)
#pragma unroll
    for (int r = 0; r < 8; ++r) {
      const int qi = q0 + 8 * g + r;
      float pv[4];
      float mx = -3.0e38f;
#pragma unroll
      for (int t = 0; t < 4; ++t) {
        float v = s[t][r] * scale;
        if (kp + t * 16 + ln > qi) v = -3.0e38f;  // pos = (kj % S) <= qi
        pv[t] = v;
        mx = fmaxf(mx, v);
      }
#pragma unroll
      for (int m = 1; m < 16; m <<= 1) mx = fmaxf(mx, __shfl_xor(mx, m, 32));
      const float nm = fmaxf(mrow[r], mx);
      const float alpha = __expf(mrow[r] - nm);
      float rs = 0.f;
#pragma unroll
      for (int t = 0; t < 4; ++t) {
        pv[t] = __expf(pv[t] - nm);
        rs += pv[t];
      }
#pragma unroll
      for (int m = 1; m < 16; m <<= 1) rs += __shfl_xor(rs, m, 32);
      lrow[r] = lrow[r] * alpha + rs;
      mrow[r] = nm;
#pragma unroll
      for (int t = 0; t < 4; ++t) o[t][r] *= alpha;
#pragma unroll
      for (int t = 0; t < 4; ++t)
        ptile[wave][8 * g + r][t * 16 + ln] = (bf16_t)pv[t];
    }

    // P (16x64) @ V (64x64): cluster P + V fragment loads, then 8 WMMAs
    // (per-wave LDS ops stay in order, so no barrier needed for ptile)
    const v16bf pf0 = load_frag(&ptile[wave][ln][0], g);
    const v16bf pf1 = load_frag(&ptile[wave][ln][32], g);
    v16bf vf[4][2];
#pragma unroll
    for (int t = 0; t < 4; ++t) {
      const bf16_t* vrow = Vt + (t * 16 + ln) * 64;
      vf[t][0] = load_frag(vrow, g);
      vf[t][1] = load_frag(vrow + 32, g);
    }
#pragma unroll
    for (int t = 0; t < 4; ++t) {
      o[t] = wmma_bf16(pf0, vf[t][0], o[t]);
      o[t] = wmma_bf16(pf1, vf[t][1], o[t]);
    }

    __syncthreads();   // all waves done with buf[cur]; safe to overwrite next iter
    cur ^= 1;
    rb = rb_n;
    kp = kp_n;
  }

  const int b = bh / H_, h = bh % H_;
#pragma unroll
  for (int t = 0; t < 4; ++t) {
#pragma unroll
    for (int r = 0; r < 8; ++r) {
      const float val = o[t][r] / lrow[r];
      const size_t sq = q0 + 8 * g + r;
      obuf[((size_t)b * S_ + sq) * D_ + (size_t)h * HD_ + t * 16 + ln] =
          (bf16_t)val;
    }
  }
}

// ---------- orchestration ----------

extern "C" void kernel_launch(void* const* d_in, const int* in_sizes, int n_in,
                              void* d_out, int out_size, void* d_ws,
                              size_t ws_size, hipStream_t stream) {
  (void)in_sizes; (void)n_in; (void)out_size; (void)ws_size;
  const float* query = (const float*)d_in[0];
  const float* context = (const float*)d_in[1];
  const float* wq = (const float*)d_in[2];
  const float* wk = (const float*)d_in[3];
  const float* wv = (const float*)d_in[4];
  const float* bq = (const float*)d_in[5];
  const float* bk = (const float*)d_in[6];
  const float* bv = (const float*)d_in[7];
  const float* wo = (const float*)d_in[8];
  const float* bo = (const float*)d_in[9];
  const float* ln1g = (const float*)d_in[10];
  const float* ln1b = (const float*)d_in[11];
  const float* ln2g = (const float*)d_in[12];
  const float* ln2b = (const float*)d_in[13];
  const float* w1 = (const float*)d_in[14];
  const float* b1 = (const float*)d_in[15];
  const float* w2 = (const float*)d_in[16];
  const float* b2 = (const float*)d_in[17];

  const size_t MB = 1u << 20;
  char* ws = (char*)d_ws;
  bf16_t* wqT = (bf16_t*)(ws + 0 * MB);    // [1024][1024]
  bf16_t* wkT = (bf16_t*)(ws + 2 * MB);
  bf16_t* wvT = (bf16_t*)(ws + 4 * MB);
  bf16_t* woT = (bf16_t*)(ws + 6 * MB);
  bf16_t* w1T = (bf16_t*)(ws + 8 * MB);    // [4096][1024]
  bf16_t* w2T = (bf16_t*)(ws + 16 * MB);   // [1024][4096]
  bf16_t* qln = (bf16_t*)(ws + 24 * MB);   // [2048][1024]
  bf16_t* ctx = (bf16_t*)(ws + 28 * MB);   // [8192][1024]
  bf16_t* qhd = (bf16_t*)(ws + 44 * MB);   // [B][H][S][64]
  bf16_t* ktb = (bf16_t*)(ws + 48 * MB);   // [B][H][Sc][64]
  bf16_t* vtb = (bf16_t*)(ws + 64 * MB);   // [B][H][64][Sc]
  bf16_t* obuf = (bf16_t*)(ws + 80 * MB);  // [2048][1024]
  float* xbuf = (float*)(ws + 84 * MB);    // [2048][1024] f32
  bf16_t* hbuf = (bf16_t*)(ws + 92 * MB);  // [2048][1024]
  bf16_t* ffb = (bf16_t*)(ws + 96 * MB);   // [2048][4096]

  // weights -> bf16, transposed [N][K]
  transpose_w<<<dim3(32, 32), 256, 0, stream>>>(wq, wqT, D_, D_);
  transpose_w<<<dim3(32, 32), 256, 0, stream>>>(wk, wkT, D_, D_);
  transpose_w<<<dim3(32, 32), 256, 0, stream>>>(wv, wvT, D_, D_);
  transpose_w<<<dim3(32, 32), 256, 0, stream>>>(wo, woT, D_, D_);
  transpose_w<<<dim3(32, 128), 256, 0, stream>>>(w1, w1T, D_, FF_);
  transpose_w<<<dim3(128, 32), 256, 0, stream>>>(w2, w2T, FF_, D_);

  // ln1(query) -> bf16; context -> bf16
  layernorm_k<<<B_ * S_, 256, 0, stream>>>(query, ln1g, ln1b, qln);
  convert_bf16<<<4096, 256, 0, stream>>>(context, ctx, (size_t)B_ * SC_ * D_);

  // projections
  gemm_nt<0><<<dim3(16, 8), 256, 0, stream>>>(qln, wqT, bq, nullptr, qhd,
                                              B_ * S_, D_, D_, S_);
  gemm_nt<0><<<dim3(64, 8), 256, 0, stream>>>(ctx, wkT, bk, nullptr, ktb,
                                              B_ * SC_, D_, D_, SC_);
  gemm_nt<1><<<dim3(64, 8), 256, 0, stream>>>(ctx, wvT, bv, nullptr, vtb,
                                              B_ * SC_, D_, D_, SC_);

  // attention
  attn_kernel<<<B_ * H_ * (S_ / 64), 128, 0, stream>>>(qhd, ktb, vtb, obuf);

  // output projection + residual -> x (f32)
  gemm_nt<2><<<dim3(16, 8), 256, 0, stream>>>(obuf, woT, bo, query, xbuf,
                                              B_ * S_, D_, D_, 0);
  // ln2(x) -> bf16
  layernorm_k<<<B_ * S_, 256, 0, stream>>>(xbuf, ln2g, ln2b, hbuf);
  // FFN up + GELU -> bf16
  gemm_nt<3><<<dim3(16, 32), 256, 0, stream>>>(hbuf, w1T, b1, nullptr, ffb,
                                               B_ * S_, FF_, D_, 0);
  // FFN down + residual x -> d_out (f32)
  gemm_nt<4><<<dim3(16, 8), 256, 0, stream>>>(ffb, w2T, b2, xbuf,
                                              (float*)d_out, B_ * S_, D_, FF_,
                                              0);
}
